// Classifier_31542239822453
// MI455X (gfx1250) — matrix operand order repaired and checked by reading, not compiled, not run on hardware
//
#include <hip/hip_runtime.h>
#include <hip/hip_bf16.h>

// ---- WMMA vector types (CDNA5 / gfx1250, wave32) ----
typedef __attribute__((ext_vector_type(16))) __bf16 v16bf;
typedef __attribute__((ext_vector_type(8)))  float  v8f;

#define S_SCALE 5.0f
#define MARGIN  0.5f
#define EPS_C   1e-7f

#define M_TILE     256   // rows (batch) per workgroup
#define N_TILE     128   // classes per workgroup
#define K_CHUNK    64    // K staged per LDS round
#define ROW_STRIDE 72    // bf16 elems per LDS row (64 data + 8 pad, 144B keeps 16B align)

__device__ __forceinline__ unsigned int pack_bf16(float a, float b) {
  unsigned int ua = __float_as_uint(a);
  unsigned int ub = __float_as_uint(b);
  ua = (ua + 0x7FFFu + ((ua >> 16) & 1u)) >> 16;   // round-to-nearest-even
  ub = (ub + 0x7FFFu + ((ub >> 16) & 1u)) >> 16;
  return ua | (ub << 16);
}

// A-matrix fragment (16x32 bf16): per lane two contiguous 8-elem K runs:
// lanes 0-15: K = k0..k0+7 and k0+16..k0+23 (k0 has +8 for lanes 16-31 folded by caller)
__device__ __forceinline__ v16bf load_frag_a(const unsigned short* row, int k0) {
  union { v16bf v; uint4 q[2]; } u;
  u.q[0] = *(const uint4*)(row + k0);
  u.q[1] = *(const uint4*)(row + k0 + 16);
  return u.v;
}
// B-matrix fragment (32x16 bf16): per lane two contiguous 8-elem K runs (k0..k0+7, k0+16..k0+23)
__device__ __forceinline__ v16bf load_frag_b(const unsigned short* row, int k0) {
  union { v16bf v; uint4 q[2]; } u;
  u.q[0] = *(const uint4*)(row + k0);
  u.q[1] = *(const uint4*)(row + k0 + 16);
  return u.v;
}

// Fused GEMM (bf16 WMMA, f32 accum) + per-row sum(exp(S*wf)).
// Writes deterministic partial sums: partials[nblk * B + row].
__global__ __launch_bounds__(256)
void gemm_expsum_kernel(const float* __restrict__ X, const float* __restrict__ W,
                        float* __restrict__ partials, int B, int D, int C) {
  __shared__ __align__(16) unsigned short As[M_TILE * ROW_STRIDE];
  __shared__ __align__(16) unsigned short Bs[N_TILE * ROW_STRIDE];
  __shared__ float rowsum[M_TILE];

  const int tid  = threadIdx.x;
  const int lane = tid & 31;
  const int wave = tid >> 5;      // 8 waves
  const int wm   = wave >> 1;     // 0..3 -> 64-row band
  const int wn   = wave & 1;      // 0..1 -> 64-col band
  const int lr   = lane & 15;
  const int half = lane >> 4;

  const int m_base = blockIdx.y * M_TILE;
  const int n_base = blockIdx.x * N_TILE;

  if (tid < M_TILE) rowsum[tid] = 0.0f;

  v8f acc[4][4] = {};

  const int rp = tid >> 4;   // row-in-pass (16 rows staged per pass)
  const int fq = tid & 15;   // float4 index within 64-col chunk

  for (int kc = 0; kc < D; kc += K_CHUNK) {
    __syncthreads();
    // ---- stage A tile: M_TILE x K_CHUNK fp32 -> bf16 LDS (coalesced b128 loads)
    #pragma unroll
    for (int pass = 0; pass < M_TILE / 16; ++pass) {
      const int r = pass * 16 + rp;
      const float4 xv = *(const float4*)(X + (size_t)(m_base + r) * D + kc + fq * 4);
      uint2 p;
      p.x = pack_bf16(xv.x, xv.y);
      p.y = pack_bf16(xv.z, xv.w);
      *(uint2*)(As + r * ROW_STRIDE + fq * 4) = p;
    }
    // ---- stage B tile: N_TILE x K_CHUNK (class rows, clamped at tail)
    #pragma unroll
    for (int pass = 0; pass < N_TILE / 16; ++pass) {
      const int r = pass * 16 + rp;
      int cls = n_base + r; if (cls >= C) cls = C - 1;
      const float4 wv = *(const float4*)(W + (size_t)cls * D + kc + fq * 4);
      uint2 p;
      p.x = pack_bf16(wv.x, wv.y);
      p.y = pack_bf16(wv.z, wv.w);
      *(uint2*)(Bs + r * ROW_STRIDE + fq * 4) = p;
    }
    // ---- prefetch next W chunk into cache hierarchy (global_prefetch_b8)
    if (kc + K_CHUNK < D) {
      int cls = n_base + (tid >> 1); if (cls >= C) cls = C - 1;
      __builtin_prefetch(W + (size_t)cls * D + kc + K_CHUNK, 0, 1);
    }
    __syncthreads();

    // ---- WMMA inner loop: two k=32 steps per staged chunk
    #pragma unroll
    for (int ks = 0; ks < K_CHUNK; ks += 32) {
      v16bf af[4], bfr[4];
      const int ka = ks + half * 8;    // A: lanes>=16 start at K+8
      const int kb = ks + half * 8;    // B: lanes>=16 start at K+8
      #pragma unroll
      for (int mt = 0; mt < 4; ++mt)
        af[mt] = load_frag_a(As + (wm * 64 + mt * 16 + lr) * ROW_STRIDE, ka);
      #pragma unroll
      for (int nt = 0; nt < 4; ++nt)
        bfr[nt] = load_frag_b(Bs + (wn * 64 + nt * 16 + lr) * ROW_STRIDE, kb);
      #pragma unroll
      for (int mt = 0; mt < 4; ++mt)
        #pragma unroll
        for (int nt = 0; nt < 4; ++nt)
          acc[mt][nt] = __builtin_amdgcn_wmma_f32_16x16x32_bf16(
              false, af[mt], false, bfr[nt], (short)0, acc[mt][nt], false, false);
    }
  }
  __syncthreads();

  // ---- epilogue: masked exp(S*wf), reduce over N within the wave's 64 cols
  #pragma unroll
  for (int mt = 0; mt < 4; ++mt) {
    #pragma unroll
    for (int r = 0; r < 8; ++r) {
      float rs = 0.0f;
      #pragma unroll
      for (int nt = 0; nt < 4; ++nt) {
        const int col = n_base + wn * 64 + nt * 16 + lr;
        rs += (col < C) ? __expf(S_SCALE * acc[mt][nt][r]) : 0.0f;
      }
      // reduce across the 16 lanes holding this row's N values
      rs += __shfl_xor(rs, 1);
      rs += __shfl_xor(rs, 2);
      rs += __shfl_xor(rs, 4);
      rs += __shfl_xor(rs, 8);
      if (lr == 0) {
        const int rowl = wm * 64 + mt * 16 + r + half * 8;
        atomicAdd(&rowsum[rowl], rs);   // exactly 2 contributors -> deterministic
      }
    }
  }
  __syncthreads();
  if (tid < M_TILE)
    partials[(size_t)blockIdx.x * B + (m_base + tid)] = rowsum[tid];
}

// diag[i] = dot(X[i], W[labels[i]]) in fp32; one wave per row
__global__ __launch_bounds__(256)
void diag_kernel(const float* __restrict__ X, const float* __restrict__ W,
                 const int* __restrict__ labels, float* __restrict__ diag,
                 int B, int D) {
  const int wave = threadIdx.x >> 5;
  const int lane = threadIdx.x & 31;
  const int row  = blockIdx.x * 8 + wave;
  if (row >= B) return;
  const int cls = labels[row];
  const float* x = X + (size_t)row * D;
  const float* w = W + (size_t)cls * D;
  float s = 0.0f;
  for (int j = lane; j < D; j += 32) s += x[j] * w[j];
  s += __shfl_xor(s, 16);
  s += __shfl_xor(s, 8);
  s += __shfl_xor(s, 4);
  s += __shfl_xor(s, 2);
  s += __shfl_xor(s, 1);
  if (lane == 0) diag[row] = s;
}

// Deterministic reduction over nblocks + ArcFace loss math + mean.
__global__ __launch_bounds__(256)
void finalize_kernel(const float* __restrict__ partials, const float* __restrict__ diag,
                     float* __restrict__ out, int nblk, int B) {
  __shared__ float red[256];
  float accv = 0.0f;
  for (int row = threadIdx.x; row < B; row += 256) {
    float rsum = 0.0f;
    for (int n = 0; n < nblk; ++n) rsum += partials[(size_t)n * B + row];
    const float d  = diag[row];
    const float cl = fminf(fmaxf(d, -1.0f + EPS_C), 1.0f - EPS_C);
    const float numer = S_SCALE * cosf(acosf(cl) + MARGIN);
    const float excl  = rsum - expf(S_SCALE * d);
    const float L = numer - logf(expf(numer) + excl);
    accv += -L;
  }
  red[threadIdx.x] = accv;
  __syncthreads();
  for (int s = 128; s > 0; s >>= 1) {
    if (threadIdx.x < s) red[threadIdx.x] += red[threadIdx.x + s];
    __syncthreads();
  }
  if (threadIdx.x == 0) out[0] = red[0] / (float)B;
}

extern "C" void kernel_launch(void* const* d_in, const int* in_sizes, int n_in,
                              void* d_out, int out_size, void* d_ws, size_t ws_size,
                              hipStream_t stream) {
  const float* X      = (const float*)d_in[0];
  const float* W      = (const float*)d_in[1];
  const int*   labels = (const int*)d_in[2];
  float*       out    = (float*)d_out;

  const int B = in_sizes[2];            // 512
  const int D = in_sizes[0] / B;        // 512
  const int C = in_sizes[1] / D;        // 100000

  const int nblk = (C + N_TILE - 1) / N_TILE;   // 782
  const int mblk = B / M_TILE;                  // 2

  float* partials = (float*)d_ws;                       // nblk * B floats
  float* diag     = partials + (size_t)nblk * B;        // B floats

  dim3 grid(nblk, mblk);
  gemm_expsum_kernel<<<grid, 256, 0, stream>>>(X, W, partials, B, D, C);
  diag_kernel<<<(B + 7) / 8, 256, 0, stream>>>(X, W, labels, diag, B, D);
  finalize_kernel<<<1, 256, 0, stream>>>(partials, diag, out, nblk, B);
}
